// SpectralConv1d_90546500534459
// MI455X (gfx1250) — matrix operand order repaired and checked by reading, not compiled, not run on hardware
//
#include <hip/hip_runtime.h>
#include <hip/hip_bf16.h>

// ---------------------------------------------------------------------------
// SpectralConv1d (FNO layer) for gfx1250 / MI455X
//   B=64, Cin=64, Cout=64, MODES=16, RES=8192
// Pipeline:
//   K0: precompute trig basis tables (bf16 hi/lo, pre-packed in WMMA B layout)
//   K1: forward truncated DFT  (GEMM 4096x32x8192) via v_wmma_f32_16x16x32_bf16
//   K2: per-mode complex channel mixing + irfft coefficient prep (VALU, tiny)
//   K3: inverse transform      (GEMM 4096x8192x32) via v_wmma_f32_16x16x32_bf16
// Precision: split-bf16 (a_hi*b_hi + a_hi*b_lo + a_lo*b_hi), f32 accumulate.
// Streams (x in, y out: 128 MB each, touched once) use non-temporal hints so
// the reused trig tables / intermediates stay resident in the 192 MB L2.
// ---------------------------------------------------------------------------

typedef __attribute__((ext_vector_type(16))) __bf16 v16bf;
typedef __attribute__((ext_vector_type(8)))  float  v8f;

#define WMMA_BF16(A, Bm, C) \
  __builtin_amdgcn_wmma_f32_16x16x32_bf16(false, (A), false, (Bm), (short)0, (C), false, false)

constexpr int kB      = 64;
constexpr int kCin    = 64;
constexpr int kCout   = 64;
constexpr int kModes  = 16;
constexpr int kRes    = 8192;
constexpr int kRows   = kB * kCin;    // 4096 DFT rows
constexpr int kNF     = 2 * kModes;   // 32 spectral coefs per row

constexpr float kTwoPiOverN = 6.283185307179586f / 8192.0f;

// Workspace layout (bytes):
//   [0,      512K) fwd twiddle hi   (256 chunks x {cos,sin} x 32 lanes x 16 bf16)
//   [512K,  1024K) fwd twiddle lo
//   [1024K, 1536K) inv basis  hi   (512 coltiles x 32 lanes x 16 bf16)
//   [1536K, 2048K) inv basis  lo
//   [2048K, 2560K) X  spectral     (4096 x 32 f32)
//   [2560K, 3072K) coef            (4096 x 32 f32)

// --- K0: trig tables, pre-packed in the WMMA B-operand lane layout ----------
// B operand (16-bit, 32x16): lane l -> column n=l&15, half h=l>>4,
//   element i -> K = 8h+i (i<8), 16+8h+(i-8) (i>=8)   (ISA 7.12.4/7.12.5 B-matrix pattern)
__global__ __launch_bounds__(256) void fno_tables(
    __bf16* __restrict__ fwd_hi, __bf16* __restrict__ fwd_lo,
    __bf16* __restrict__ inv_hi, __bf16* __restrict__ inv_lo) {
  int tid = blockIdx.x * 256 + threadIdx.x;   // 2048 blocks -> 524288 threads
  if (tid < 262144) {
    // forward: idx = ((kc*2 + t)*32 + lane)*16 + i ; t: 0=cos, 1=-sin
    int i  = tid & 15;
    int l  = (tid >> 4) & 31;
    int t  = (tid >> 9) & 1;
    int kc = tid >> 10;
    int m  = l & 15;
    int h  = l >> 4;
    int K  = (i < 8) ? (h * 8 + i) : (16 + h * 8 + (i - 8));
    int k  = kc * 32 + K;
    int ph = (m * k) & (kRes - 1);
    float s, c;
    __sincosf((float)ph * kTwoPiOverN, &s, &c);
    float v = t ? -s : c;
    __bf16 hi = (__bf16)v;
    fwd_hi[tid] = hi;
    fwd_lo[tid] = (__bf16)(v - (float)hi);
  } else {
    // inverse: idx = (ct*32 + lane)*16 + i ; basis row K: even->cos(m n), odd->sin(m n)
    int u  = tid - 262144;
    int i  = u & 15;
    int l  = (u >> 4) & 31;
    int ct = u >> 9;
    int h  = l >> 4;
    int K  = (i < 8) ? (h * 8 + i) : (16 + h * 8 + (i - 8));
    int m  = K >> 1;
    int n  = ct * 16 + (l & 15);
    int ph = (m * n) & (kRes - 1);
    float s, c;
    __sincosf((float)ph * kTwoPiOverN, &s, &c);
    float v = (K & 1) ? s : c;
    __bf16 hi = (__bf16)v;
    inv_hi[u] = hi;
    inv_lo[u] = (__bf16)(v - (float)hi);
  }
}

// --- K1: forward truncated DFT ---------------------------------------------
// Grid: 256 blocks (one 16-row tile each), 256 threads = 8 waves.
// Each wave covers a contiguous 1 KB K-range per row (32 chunks of 32),
// accumulating a 16x32 f32 tile; partials reduced through LDS.
// x is streamed once -> non-temporal loads (one full 128 B line/row/chunk).
__global__ __launch_bounds__(256) void fno_fwd_dft(
    const float* __restrict__ x,
    const __bf16* __restrict__ tw_hi, const __bf16* __restrict__ tw_lo,
    float* __restrict__ Xf) {
  __shared__ float red[8 * 512];
  const int tid   = threadIdx.x;
  const int wave  = tid >> 5;
  const int lane  = tid & 31;
  const int m16   = lane & 15;
  const int h     = lane >> 4;
  const int rbase = blockIdx.x * 16;

  const float* xrow = x + (size_t)(rbase + m16) * kRes;

  v8f accC, accS;
#pragma unroll
  for (int j = 0; j < 8; ++j) { accC[j] = 0.0f; accS[j] = 0.0f; }

  for (int it = 0; it < 32; ++it) {
    const int kc = wave * 32 + it;
    const int k0 = kc * 32;
    // A operand (16-bit 16x32): lane half h -> K = {8h+i (i<8), 16+8h+(i-8)}
    const float* p = xrow + k0 + h * 8;
    float xv[16];
#pragma unroll
    for (int j = 0; j < 8; ++j) xv[j] = __builtin_nontemporal_load(p + j);
#pragma unroll
    for (int j = 0; j < 8; ++j) xv[8 + j] = __builtin_nontemporal_load(p + 16 + j);

    v16bf a_hi, a_lo;
#pragma unroll
    for (int j = 0; j < 16; ++j) {
      float v = xv[j];
      __bf16 hi = (__bf16)v;
      a_hi[j] = hi;
      a_lo[j] = (__bf16)(v - (float)hi);
    }

    const size_t tb = ((size_t)(kc * 2) * 32 + lane) * 16;
    v16bf bc_hi = *(const v16bf*)(tw_hi + tb);
    v16bf bc_lo = *(const v16bf*)(tw_lo + tb);
    v16bf bs_hi = *(const v16bf*)(tw_hi + tb + 512);
    v16bf bs_lo = *(const v16bf*)(tw_lo + tb + 512);

    accC = WMMA_BF16(a_hi, bc_hi, accC);
    accC = WMMA_BF16(a_hi, bc_lo, accC);
    accC = WMMA_BF16(a_lo, bc_hi, accC);
    accS = WMMA_BF16(a_hi, bs_hi, accS);
    accS = WMMA_BF16(a_hi, bs_lo, accS);
    accS = WMMA_BF16(a_lo, bs_hi, accS);
  }

  // C/D layout: VGPR j -> row M = j + 8h, col = m16
  const int base = wave * 512;
#pragma unroll
  for (int j = 0; j < 8; ++j) {
    const int M = j + 8 * h;
    red[base + M * 32 + m16]      = accC[j];
    red[base + M * 32 + 16 + m16] = accS[j];
  }
  __syncthreads();

  for (int e = tid; e < 512; e += 256) {
    float s = 0.0f;
#pragma unroll
    for (int w = 0; w < 8; ++w) s += red[w * 512 + e];
    Xf[(size_t)rbase * kNF + e] = s;   // row = rbase + e/32, col = e%32
  }
}

// --- K2: complex channel mixing + irfft coefficient prep -------------------
// A[b,o,m] = sum_i X[b,i,m] * W[i,o,m]   (complex)
// coef[r,2m]   = (m==0 ? ReA : 2 ReA)/N          (cos weight)
// coef[r,2m+1] = (m==0 ? 0   : -2 ImA)/N         (sin weight)
__global__ __launch_bounds__(256) void fno_mix(
    const float* __restrict__ Xf, const float* __restrict__ wr,
    const float* __restrict__ wi, float* __restrict__ coef) {
  const int t = blockIdx.x * 256 + threadIdx.x;  // 65536 = B*Cout*MODES
  const int m = t & 15;
  const int o = (t >> 4) & 63;
  const int b = t >> 10;

  const float* xb = Xf + (size_t)b * kCin * kNF;
  float re = 0.0f, im = 0.0f;
#pragma unroll 4
  for (int i = 0; i < kCin; ++i) {
    const float xr  = xb[i * kNF + m];
    const float xi  = xb[i * kNF + 16 + m];
    const float wrv = wr[(i * kCout + o) * kModes + m];
    const float wiv = wi[(i * kCout + o) * kModes + m];
    re += xr * wrv - xi * wiv;
    im += xr * wiv + xi * wrv;
  }
  const float invN = 1.0f / (float)kRes;
  const float cc = (m == 0 ? re : 2.0f * re) * invN;
  const float cs = (m == 0 ? 0.0f : -2.0f * im) * invN;
  const size_t r = (size_t)b * kCout + o;
  coef[r * kNF + 2 * m]     = cc;
  coef[r * kNF + 2 * m + 1] = cs;
}

// --- K3: inverse transform --------------------------------------------------
// Grid: (256 row tiles, 8 col groups), 256 threads = 8 waves.
// Each wave: fixed 16-row coef tile (A operand, split once), then a
// CONTIGUOUS run of 8 column tiles so consecutive iterations fill
// consecutive 64 B segments of the same output cache lines.
// y is written once -> non-temporal stores.
__global__ __launch_bounds__(256) void fno_idft(
    const float* __restrict__ coef,
    const __bf16* __restrict__ gb_hi, const __bf16* __restrict__ gb_lo,
    float* __restrict__ out) {
  const int tid   = threadIdx.x;
  const int wave  = tid >> 5;
  const int lane  = tid & 31;
  const int m16   = lane & 15;
  const int h     = lane >> 4;
  const int rbase = blockIdx.x * 16;
  const int cg    = blockIdx.y;

  // A operand: lane half h -> K = {8h+i (i<8), 16+8h+(i-8)}
  const float* crow = coef + (size_t)(rbase + m16) * kNF + h * 8;
  v16bf a_hi, a_lo;
#pragma unroll
  for (int j = 0; j < 16; ++j) {
    const float v = (j < 8) ? crow[j] : crow[j + 8];
    __bf16 hi = (__bf16)v;
    a_hi[j] = hi;
    a_lo[j] = (__bf16)(v - (float)hi);
  }

  const int ct0 = cg * 64 + wave * 8;   // contiguous 8-tile run per wave
#pragma unroll 2
  for (int ct = ct0; ct < ct0 + 8; ++ct) {
    const size_t tb = ((size_t)ct * 32 + lane) * 16;
    v16bf b_hi = *(const v16bf*)(gb_hi + tb);
    v16bf b_lo = *(const v16bf*)(gb_lo + tb);

    v8f acc;
#pragma unroll
    for (int j = 0; j < 8; ++j) acc[j] = 0.0f;
    acc = WMMA_BF16(a_hi, b_hi, acc);
    acc = WMMA_BF16(a_hi, b_lo, acc);
    acc = WMMA_BF16(a_lo, b_hi, acc);

    const int n0 = ct * 16;
    float* orow = out + (size_t)(rbase + 8 * h) * kRes + n0 + m16;
#pragma unroll
    for (int j = 0; j < 8; ++j)
      __builtin_nontemporal_store(acc[j], orow + (size_t)j * kRes);
  }
}

// ---------------------------------------------------------------------------
extern "C" void kernel_launch(void* const* d_in, const int* in_sizes, int n_in,
                              void* d_out, int out_size, void* d_ws, size_t ws_size,
                              hipStream_t stream) {
  (void)in_sizes; (void)n_in; (void)out_size; (void)ws_size;
  const float* x  = (const float*)d_in[0];
  const float* wr = (const float*)d_in[1];
  const float* wi = (const float*)d_in[2];
  float* out = (float*)d_out;

  char* ws = (char*)d_ws;
  __bf16* fwd_hi = (__bf16*)(ws);
  __bf16* fwd_lo = (__bf16*)(ws + (512u << 10));
  __bf16* inv_hi = (__bf16*)(ws + (1024u << 10));
  __bf16* inv_lo = (__bf16*)(ws + (1536u << 10));
  float*  Xf     = (float*)(ws + (2048u << 10));
  float*  coef   = (float*)(ws + (2560u << 10));

  fno_tables<<<2048, 256, 0, stream>>>(fwd_hi, fwd_lo, inv_hi, inv_lo);
  fno_fwd_dft<<<kRows / 16, 256, 0, stream>>>(x, fwd_hi, fwd_lo, Xf);
  fno_mix<<<(kB * kCout * kModes) / 256, 256, 0, stream>>>(Xf, wr, wi, coef);
  fno_idft<<<dim3(kRows / 16, 8), 256, 0, stream>>>(coef, inv_hi, inv_lo, out);
}
